// HeteroGraphClassifier_7086696038635
// MI455X (gfx1250) — matrix operand order repaired and checked by reading, not compiled, not run on hardware
//
#include <hip/hip_runtime.h>
#include <hip/hip_bf16.h>

typedef __attribute__((ext_vector_type(16))) _Float16 v16h;
typedef __attribute__((ext_vector_type(8)))  float    v8f;

#define NNODES 100000
#define HDIM   256
#define NGRAPH 128
#define NEDGE  500000

union HFrag { v16h v; _Float16 h[16]; };
union FFrag { v8f  v; float    f[8]; };

// ---------------------------------------------------------------------------
// Pack one 256x256 f32 weight matrix (row = k, col = n) into WMMA B-fragment
// order for v_wmma_f32_16x16x32_f16:
//   flat p = ((kstep*16 + ntile)*32 + lane)*16 + j
//   n = ntile*16 + (lane&15);  k = kstep*32 + ((lane>>4)<<3) + j + (j>=8 ? 8 : 0)
// Each lane's 16 f16 (32B) are then contiguous -> two global_load_b128.
// ---------------------------------------------------------------------------
__global__ __launch_bounds__(256) void packw_kernel(const float* __restrict__ W,
                                                    _Float16* __restrict__ Wp) {
    int p = blockIdx.x * 256 + threadIdx.x;          // 0..65535
    int j    =  p        & 15;
    int lane = (p >> 4)  & 31;
    int nt   = (p >> 9)  & 15;
    int ks   =  p >> 13;
    int n = nt * 16 + (lane & 15);
    int k = ks * 32 + ((lane >> 4) << 3) + j + ((j >= 8) ? 8 : 0);
    Wp[p] = (_Float16)W[k * HDIM + n];
}

// ---------------------------------------------------------------------------
// Node encoder: h = cate_emb[cate] + level_emb[level] + sinusoidal_pe[thought]
// ---------------------------------------------------------------------------
__global__ __launch_bounds__(256) void encode_kernel(const int* __restrict__ x,
                                                     const float* __restrict__ cate_emb,
                                                     const float* __restrict__ level_emb,
                                                     float* __restrict__ h) {
    int node = blockIdx.x;
    int c    = threadIdx.x;
    int level   = min(max(x[node * 3 + 0], 0), 999);
    int cate    = min(max(x[node * 3 + 1], 0), 4);
    int thought = min(max(x[node * 3 + 2], 0), 4999);
    // pe column c uses divisor exp(-(c&~1) * ln(10000)/256); even->sin, odd->cos
    float div = __expf(-(float)(c & ~1) * (9.210340371976184f / 256.0f));
    float ang = (float)thought * div;
    float pe  = (c & 1) ? __cosf(ang) : __sinf(ang);
    h[(size_t)node * HDIM + c] = cate_emb[cate * HDIM + c] + level_emb[level * HDIM + c] + pe;
}

// ---------------------------------------------------------------------------
// Edge scatter: z[dst] += h[src]  (atomic f32, 64 threads x float4 per edge)
// ---------------------------------------------------------------------------
__global__ __launch_bounds__(64) void scatter_add_kernel(const int* __restrict__ ei,
                                                         const float* __restrict__ h,
                                                         float* __restrict__ z) {
    int e   = blockIdx.x;
    int src = ei[e];
    int dst = ei[NEDGE + e];
    int c   = threadIdx.x * 4;
    const float4 v = *(const float4*)&h[(size_t)src * HDIM + c];
    float* zp = &z[(size_t)dst * HDIM + c];
    atomicAdd(zp + 0, v.x);
    atomicAdd(zp + 1, v.y);
    atomicAdd(zp + 2, v.z);
    atomicAdd(zp + 3, v.w);
}

// ---------------------------------------------------------------------------
// WMMA GEMM: C[M,256] (+)= op(A[M,256]) @ B[256,256] + bias
//   mode bit0: accumulate into C (else store)
//   mode bit1: A-transform relu(A*bnA[k] + bnC[k]) during f32->f16 staging
//   mode bit2: accumulate per-column sum/sumsq of result into stats[2][256]
// Block: 256 threads = 8 waves; tile 64 rows x 256 cols; each wave: 16x128,
// 64 x v_wmma_f32_16x16x32_f16. A fragments staged in LDS in fragment order.
// ---------------------------------------------------------------------------
__global__ __launch_bounds__(256) void gemm256_wmma(
    const float* __restrict__ A, const _Float16* __restrict__ Bp,
    const float* __restrict__ bias, float* __restrict__ C,
    float* __restrict__ stats, const float* __restrict__ bnA,
    const float* __restrict__ bnC, int M, int mode)
{
    __shared__ v16h  As[1024];        // [rg(4)][kstep(8)][lane(32)] = 32 KB
    __shared__ float sstat[512];
    const int tid  = threadIdx.x;
    const int row0 = blockIdx.x * 64;

    // Stage A in WMMA A-fragment order (ISA 16-bit A 16x32 layout):
    //   m = lane&15;  k = kstep*32 + (lane>>4)*8 + (j<8 ? j : j+8)
    for (int trip = tid; trip < 1024; trip += 256) {
        const int lane  =  trip       & 31;
        const int kstep = (trip >> 5) & 7;
        const int rg    =  trip >> 8;
        const int m  = row0 + rg * 16 + (lane & 15);
        const int kb = kstep * 32 + ((lane >> 4) << 3);
        HFrag f;
        if (m < M) {
            const float* src = A + (size_t)m * HDIM;
#pragma unroll
            for (int j = 0; j < 16; ++j) {
                const int k = kb + j + ((j >= 8) ? 8 : 0);
                float v = src[k];
                if (mode & 2) v = fmaxf(v * bnA[k] + bnC[k], 0.f);
                f.h[j] = (_Float16)v;
            }
        } else {
#pragma unroll
            for (int j = 0; j < 16; ++j) f.h[j] = (_Float16)0.f;
        }
        As[trip] = f.v;
    }
    if (mode & 4)
        for (int i = tid; i < 512; i += 256) sstat[i] = 0.f;
    __syncthreads();

    const int lane = tid & 31;
    const int wave = tid >> 5;
    const int rg   = wave & 3;   // row group (16 rows)
    const int cg   = wave >> 2;  // column half (128 cols)

    v8f acc[8] = {};
#pragma unroll
    for (int kstep = 0; kstep < 8; ++kstep) {
        const v16h a = As[(rg * 8 + kstep) * 32 + lane];
        const _Float16* bp = Bp + ((size_t)((kstep * 16 + cg * 8) * 32 + lane)) * 16;
#pragma unroll
        for (int nt = 0; nt < 8; ++nt) {
            const v16h b = *(const v16h*)(bp + (size_t)nt * 512);
            acc[nt] = __builtin_amdgcn_wmma_f32_16x16x32_f16(
                false, a, false, b, (short)0, acc[nt], false, false);
        }
    }

    // C 16x16 f32 layout: VGPR v -> row v + (lane>=16 ? 8 : 0); col = lane&15
    const int mrow = row0 + rg * 16 + ((lane >> 4) << 3);
#pragma unroll
    for (int nt = 0; nt < 8; ++nt) {
        const int n  = cg * 128 + nt * 16 + (lane & 15);
        const float bi = bias[n];
        FFrag u; u.v = acc[nt];
        float s = 0.f, sq = 0.f;
#pragma unroll
        for (int v = 0; v < 8; ++v) {
            const int m = mrow + v;
            if (m < M) {
                float val = u.f[v] + bi;
                float* cp = C + (size_t)m * HDIM + n;
                if (mode & 1) val += *cp;
                *cp = val;
                s += val; sq += val * val;
            }
        }
        if (mode & 4) { atomicAdd(&sstat[n], s); atomicAdd(&sstat[256 + n], sq); }
    }
    __syncthreads();
    if (mode & 4) {
        atomicAdd(&stats[tid],       sstat[tid]);
        atomicAdd(&stats[256 + tid], sstat[256 + tid]);
    }
}

// ---------------------------------------------------------------------------
// BN fold: a = g*rsqrt(var+eps), c = beta - mean*a (biased stats over `count`)
// ---------------------------------------------------------------------------
__global__ __launch_bounds__(256) void bnparam_kernel(const float* __restrict__ stats,
                                                      const float* __restrict__ g,
                                                      const float* __restrict__ beta,
                                                      float* __restrict__ bnA,
                                                      float* __restrict__ bnC,
                                                      float count) {
    int c = threadIdx.x;
    float mean = stats[c] / count;
    float var  = stats[256 + c] / count - mean * mean;
    float a    = g[c] * rsqrtf(var + 1e-5f);
    bnA[c] = a;
    bnC[c] = beta[c] - mean * a;
}

// Column sums / sums of squares of X[M,256] into stats[2][256]
__global__ __launch_bounds__(256) void colstats_kernel(const float* __restrict__ X,
                                                       float* __restrict__ stats, int M) {
    int c  = threadIdx.x;
    int r0 = blockIdx.x * 128;
    int r1 = min(r0 + 128, M);
    float s = 0.f, sq = 0.f;
    for (int r = r0; r < r1; ++r) {
        float v = X[(size_t)r * HDIM + c];
        s += v; sq += v * v;
    }
    atomicAdd(&stats[c], s);
    atomicAdd(&stats[256 + c], sq);
}

__global__ __launch_bounds__(256) void bnrelu_kernel(const float* __restrict__ X,
                                                     const float* __restrict__ bnA,
                                                     const float* __restrict__ bnC,
                                                     float* __restrict__ Y, size_t total) {
    size_t stride = (size_t)gridDim.x * blockDim.x;
    for (size_t i = (size_t)blockIdx.x * blockDim.x + threadIdx.x; i < total; i += stride) {
        int c = (int)(i & (HDIM - 1));
        Y[i] = fmaxf(X[i] * bnA[c] + bnC[c], 0.f);
    }
}

__global__ __launch_bounds__(256) void zero_kernel(float* __restrict__ p, int n) {
    int stride = gridDim.x * blockDim.x;
    for (int i = blockIdx.x * blockDim.x + threadIdx.x; i < n; i += stride) p[i] = 0.f;
}

// Global mean pool: scatter then divide
__global__ __launch_bounds__(256) void pool_scatter_kernel(const float* __restrict__ h,
                                                           const int* __restrict__ batch,
                                                           float* __restrict__ pooled,
                                                           float* __restrict__ counts) {
    int node = blockIdx.x;
    int c    = threadIdx.x;
    int g    = batch[node];
    atomicAdd(&pooled[(size_t)g * HDIM + c], h[(size_t)node * HDIM + c]);
    if (c == 0) atomicAdd(&counts[g], 1.0f);
}

__global__ __launch_bounds__(256) void pool_div_kernel(float* __restrict__ pooled,
                                                       const float* __restrict__ counts) {
    int g = blockIdx.x;
    int c = threadIdx.x;
    pooled[(size_t)g * HDIM + c] /= fmaxf(counts[g], 1.0f);
}

// Final head: out[g, 0:2] = relu(bn(y[g,:])) @ Wc2 + bc2
__global__ __launch_bounds__(256) void head2_kernel(const float* __restrict__ y,
                                                    const float* __restrict__ bnA,
                                                    const float* __restrict__ bnC,
                                                    const float* __restrict__ Wc2,
                                                    const float* __restrict__ bc2,
                                                    float* __restrict__ out) {
    __shared__ float s0[256];
    __shared__ float s1[256];
    int g = blockIdx.x;
    int k = threadIdx.x;
    float r = fmaxf(y[(size_t)g * HDIM + k] * bnA[k] + bnC[k], 0.f);
    s0[k] = r * Wc2[k * 2 + 0];
    s1[k] = r * Wc2[k * 2 + 1];
    __syncthreads();
    for (int off = 128; off > 0; off >>= 1) {
        if (k < off) { s0[k] += s0[k + off]; s1[k] += s1[k + off]; }
        __syncthreads();
    }
    if (k == 0) {
        out[g * 2 + 0] = s0[0] + bc2[0];
        out[g * 2 + 1] = s1[0] + bc2[1];
    }
}

// ---------------------------------------------------------------------------
extern "C" void kernel_launch(void* const* d_in, const int* in_sizes, int n_in,
                              void* d_out, int out_size, void* d_ws, size_t ws_size,
                              hipStream_t stream) {
    (void)in_sizes; (void)n_in; (void)out_size; (void)ws_size;

    const int*   x         = (const int*)  d_in[0];
    const int*   ei[3]     = { (const int*)d_in[1], (const int*)d_in[2], (const int*)d_in[3] };
    const int*   batch     = (const int*)  d_in[4];
    const float* cate_emb  = (const float*)d_in[5];
    const float* level_emb = (const float*)d_in[6];
    const float* gin_W1    = (const float*)d_in[7];
    const float* gin_b1    = (const float*)d_in[8];
    const float* gin_g     = (const float*)d_in[9];
    const float* gin_beta  = (const float*)d_in[10];
    const float* gin_W2    = (const float*)d_in[11];
    const float* gin_b2    = (const float*)d_in[12];
    const float* norm_g    = (const float*)d_in[13];
    const float* norm_b    = (const float*)d_in[14];
    const float* Wc1       = (const float*)d_in[15];
    const float* bc1       = (const float*)d_in[16];
    const float* cg_       = (const float*)d_in[17];
    const float* cb_       = (const float*)d_in[18];
    const float* Wc2       = (const float*)d_in[19];
    const float* bc2       = (const float*)d_in[20];
    float* out = (float*)d_out;

    const size_t NH = (size_t)NNODES * HDIM;
    float* ws     = (float*)d_ws;
    float* h      = ws;                 // [N,256]
    float* zbuf   = ws + NH;            // [N,256]
    float* ybuf   = ws + 2 * NH;        // [N,256]
    float* obuf   = ws + 3 * NH;        // [N,256]
    float* stats  = ws + 4 * NH;        // [2,256]
    float* bnA    = stats + 512;        // [256]
    float* bnC    = bnA + 256;          // [256]
    float* pooled = bnC + 256;          // [128,256]
    float* counts = pooled + (size_t)NGRAPH * HDIM;  // [128]
    _Float16* wpack = (_Float16*)(counts + NGRAPH + 128);  // 32B-aligned
    _Float16* W1p  = wpack;
    _Float16* W2p  = wpack + (size_t)9  * 65536;
    _Float16* Wc1p = wpack + (size_t)18 * 65536;

    // Pack all weight matrices to WMMA fragment order (f16)
    for (int i = 0; i < 9; ++i) {
        packw_kernel<<<256, 256, 0, stream>>>(gin_W1 + (size_t)i * 65536, W1p + (size_t)i * 65536);
        packw_kernel<<<256, 256, 0, stream>>>(gin_W2 + (size_t)i * 65536, W2p + (size_t)i * 65536);
    }
    packw_kernel<<<256, 256, 0, stream>>>(Wc1, Wc1p);

    // Node encoder
    encode_kernel<<<NNODES, 256, 0, stream>>>(x, cate_emb, level_emb, h);

    const int gblocks = (NNODES + 63) / 64;  // 1563
    for (int l = 0; l < 3; ++l) {
        for (int t = 0; t < 3; ++t) {
            const int lt = l * 3 + t;
            // z = h + segment_sum(h[src] -> dst)
            hipMemcpyAsync(zbuf, h, NH * sizeof(float), hipMemcpyDeviceToDevice, stream);
            scatter_add_kernel<<<NEDGE, 64, 0, stream>>>(ei[t], h, zbuf);
            // y = z @ W1 + b1, fused column stats for inner BN
            zero_kernel<<<2, 256, 0, stream>>>(stats, 512);
            gemm256_wmma<<<gblocks, 256, 0, stream>>>(
                zbuf, W1p + (size_t)lt * 65536, gin_b1 + lt * 256, ybuf,
                stats, bnA, bnC, NNODES, /*mode=*/4);
            bnparam_kernel<<<1, 256, 0, stream>>>(stats, gin_g + lt * 256, gin_beta + lt * 256,
                                                  bnA, bnC, (float)NNODES);
            // out (+)= relu(bn(y)) @ W2 + b2   (BN+ReLU fused into A staging)
            gemm256_wmma<<<gblocks, 256, 0, stream>>>(
                ybuf, W2p + (size_t)lt * 65536, gin_b2 + lt * 256, obuf,
                stats, bnA, bnC, NNODES, /*mode=*/2 | (t > 0 ? 1 : 0));
        }
        // outer per-layer BN + ReLU -> h
        zero_kernel<<<2, 256, 0, stream>>>(stats, 512);
        colstats_kernel<<<(NNODES + 127) / 128, 256, 0, stream>>>(obuf, stats, NNODES);
        bnparam_kernel<<<1, 256, 0, stream>>>(stats, norm_g + l * 256, norm_b + l * 256,
                                              bnA, bnC, (float)NNODES);
        bnrelu_kernel<<<4096, 256, 0, stream>>>(obuf, bnA, bnC, h, NH);
    }

    // Global mean pool
    zero_kernel<<<(NGRAPH * HDIM + NGRAPH + 255) / 256, 256, 0, stream>>>(
        pooled, NGRAPH * HDIM + NGRAPH);
    pool_scatter_kernel<<<NNODES, 256, 0, stream>>>(h, batch, pooled, counts);
    pool_div_kernel<<<NGRAPH, 256, 0, stream>>>(pooled, counts);

    // Classifier head: y = pooled @ Wc1 + bc1 (WMMA, M=128), BN, relu, @ Wc2 + bc2
    zero_kernel<<<2, 256, 0, stream>>>(stats, 512);
    gemm256_wmma<<<2, 256, 0, stream>>>(pooled, Wc1p, bc1, ybuf, stats, bnA, bnC,
                                        NGRAPH, /*mode=*/4);
    bnparam_kernel<<<1, 256, 0, stream>>>(stats, cg_, cb_, bnA, bnC, (float)NGRAPH);
    head2_kernel<<<NGRAPH, 256, 0, stream>>>(ybuf, bnA, bnC, Wc2, bc2, out);
}
